// MHAtt_68418829025594
// MI455X (gfx1250) — compile-verified, matrix operand order checked by reading, not run on hardware
//
#include <hip/hip_runtime.h>
#include <hip/hip_bf16.h>
#include <math.h>

#define NH   16
#define HD   64
#define HID  1024
#define SEQ  512
#define BATCH 4
#define EPSW 1e-7f

typedef __attribute__((ext_vector_type(16))) __bf16 bf16x16;
typedef __attribute__((ext_vector_type(8)))  float  f32x8;

// K-index mapping for 16-bit A/B fragments (ISA 7.12.2):
// lanes 0-15 hold K {0..7, 16..23}, lanes 16-31 hold K {8..15, 24..31}
__device__ __forceinline__ int kmap(int i, int half) {
    return ((i < 8) ? i : i + 8) + half * 8;
}

// Low 32 bits of a generic pointer into LDS == wave-relative LDS byte offset.
__device__ __forceinline__ unsigned lds_addr(const void* p) {
    return (unsigned)(unsigned long long)(uintptr_t)p;
}
__device__ __forceinline__ unsigned long long gaddr(const void* p) {
    return (unsigned long long)(uintptr_t)p;
}

// 16-byte async copy global -> LDS (ASYNCcnt-tracked, no VGPR staging).
// INST_OFFSET is added to both the LDS and the global address.
__device__ __forceinline__ void async_cp16(unsigned lds, unsigned long long ga) {
    asm volatile("global_load_async_to_lds_b128 %0, %1, off"
                 :: "v"(lds), "v"(ga) : "memory");
}
__device__ __forceinline__ void async_cp16_off16(unsigned lds, unsigned long long ga) {
    asm volatile("global_load_async_to_lds_b128 %0, %1, off offset:16"
                 :: "v"(lds), "v"(ga) : "memory");
}
// Async loads complete in order per wave: waiting for <= N leaves the next
// tile's N copies in flight while guaranteeing the previous tile landed.
__device__ __forceinline__ void wait_async0() {
    asm volatile("s_wait_asynccnt 0x0" ::: "memory");
}
__device__ __forceinline__ void wait_async1() {
    asm volatile("s_wait_asynccnt 0x1" ::: "memory");
}
__device__ __forceinline__ void wait_async3() {
    asm volatile("s_wait_asynccnt 0x3" ::: "memory");
}

// ------------------------- fp32 -> bf16 conversion -------------------------
__global__ void cvt_f32_bf16(const float* __restrict__ src, __bf16* __restrict__ dst, int n) {
    int i = blockIdx.x * blockDim.x + threadIdx.x;
    int stride = gridDim.x * blockDim.x;
    for (; i < n; i += stride) dst[i] = (__bf16)src[i];
}

// ------------------------- generic WMMA GEMM -------------------------------
// C[M,N] = A[M,K] * W[N,K]^T + bias[N];  M = BATCH*SEQ, K = N = HID
// Block tile 128x64, 8 waves, each wave 16x64 (4 WMMA tiles sharing one A frag).
// Double-buffered async LDS staging (3 async b128 per thread per K-tile).
// MODE 0: head-split bf16 output [B,H,S,D]   MODE 1: flat fp32 output [M,N]
template<int MODE>
__global__ void gemm_wmma(const __bf16* __restrict__ A, const __bf16* __restrict__ W,
                          const float* __restrict__ bias,
                          __bf16* __restrict__ outb, float* __restrict__ outf) {
    __shared__ __bf16 At[2][128][32];
    __shared__ __bf16 Wt[2][64][32];
    const int tid  = threadIdx.x;
    const int wv   = tid >> 5, lane = tid & 31;
    const int half = lane >> 4, ln = lane & 15;
    const int rowbase = blockIdx.y * 128;
    const int nbase   = blockIdx.x * 64;

    f32x8 acc[4];
    f32x8 z = {};
    #pragma unroll
    for (int t = 0; t < 4; ++t) acc[t] = z;

    // async staging assignments: A 128x32 bf16 = 8KB -> 32B/thread (2x b128)
    //                            W  64x32 bf16 = 4KB -> 16B/thread (1x b128)
    const int ar = tid >> 1;              // A row 0..127
    const int ac = (tid & 1) * 16;        // A col 0|16
    const int wr = tid >> 2;              // W row 0..63
    const int wc = (tid & 3) * 8;         // W col 0|8|16|24
    const unsigned a_lds[2] = { lds_addr(&At[0][ar][ac]), lds_addr(&At[1][ar][ac]) };
    const unsigned w_lds[2] = { lds_addr(&Wt[0][wr][wc]), lds_addr(&Wt[1][wr][wc]) };
    const __bf16* agp = A + (size_t)(rowbase + ar) * HID + ac;
    const __bf16* wgp = W + (size_t)(nbase + wr) * HID + wc;

    // prologue: stage K-tile 0 into buffer 0
    async_cp16      (a_lds[0], gaddr(agp));
    async_cp16_off16(a_lds[0], gaddr(agp));
    async_cp16      (w_lds[0], gaddr(wgp));

    const int NT = HID / 32;
    for (int t = 0; t < NT; ++t) {
        const int buf = t & 1;
        if (t + 1 < NT) {   // stream next tile into the other buffer
            const __bf16* an = agp + (t + 1) * 32;
            const __bf16* wn = wgp + (t + 1) * 32;
            async_cp16      (a_lds[buf ^ 1], gaddr(an));
            async_cp16_off16(a_lds[buf ^ 1], gaddr(an));
            async_cp16      (w_lds[buf ^ 1], gaddr(wn));
            wait_async3();  // tile t landed; tile t+1 still in flight
        } else {
            wait_async0();
        }
        __syncthreads();

        bf16x16 a;
        #pragma unroll
        for (int i = 0; i < 16; ++i) a[i] = At[buf][wv * 16 + ln][kmap(i, half)];
        #pragma unroll
        for (int nt = 0; nt < 4; ++nt) {
            bf16x16 b;
            #pragma unroll
            for (int i = 0; i < 16; ++i) b[i] = Wt[buf][nt * 16 + ln][kmap(i, half)];
            acc[nt] = __builtin_amdgcn_wmma_f32_16x16x32_bf16(false, a, false, b,
                                                              (short)0, acc[nt], false, false);
        }
        __syncthreads();
    }

    #pragma unroll
    for (int nt = 0; nt < 4; ++nt) {
        const int n = nbase + nt * 16 + ln;
        const float bn = bias[n];
        #pragma unroll
        for (int r = 0; r < 8; ++r) {
            int m = rowbase + wv * 16 + r + half * 8;
            float vo = acc[nt][r] + bn;
            if (MODE == 0) {
                int b_ = m / SEQ, s = m % SEQ;
                int h  = n / HD, d = n % HD;
                outb[((size_t)(b_ * NH + h) * SEQ + s) * HD + d] = (__bf16)vo;
            } else {
                outf[(size_t)m * HID + n] = vo;
            }
        }
    }
}

// ---------------- fused scores + softmax + Weibull sampling ----------------
// grid: (SEQ/16, BATCH*NH). Each block: 16 q-rows, full 512 k-columns.
__global__ void attn_weibull(const __bf16* __restrict__ q1, const __bf16* __restrict__ k1,
                             const __bf16* __restrict__ q2, const __bf16* __restrict__ k2,
                             const float* __restrict__ u, const unsigned char* __restrict__ mask,
                             float* __restrict__ kout, __bf16* __restrict__ att) {
    __shared__ float  sc1[16][SEQ];   // masked+scaled scores1
    __shared__ __bf16 kwl[16][SEQ];   // clipped Weibull shape k
    const int tid  = threadIdx.x;
    const int wv   = tid >> 5, lane = tid & 31;
    const int half = lane >> 4, ln = lane & 15;
    const int qt = blockIdx.x;
    const int bh = blockIdx.y;
    const int b  = bh >> 4;
    const int qbase = qt * 16;
    const size_t hbase = (size_t)bh * SEQ;

    const __bf16* q1p = q1 + (hbase + qbase) * HD;
    const __bf16* q2p = q2 + (hbase + qbase) * HD;

    // 32 column tiles of 16, 4 per wave
    for (int nt = wv * 4; nt < wv * 4 + 4; ++nt) {
        f32x8 acc1 = {}, acc2 = {};
        const __bf16* k1p = k1 + (hbase + nt * 16 + ln) * HD;
        const __bf16* k2p = k2 + (hbase + nt * 16 + ln) * HD;
        #pragma unroll
        for (int dstep = 0; dstep < HD; dstep += 32) {
            bf16x16 a1, a2, b1, b2;
            #pragma unroll
            for (int i = 0; i < 16; ++i) {
                int kk = dstep + kmap(i, half);
                a1[i] = q1p[(size_t)ln * HD + kk];
                a2[i] = q2p[(size_t)ln * HD + kk];
                b1[i] = k1p[kk];
                b2[i] = k2p[kk];
            }
            acc1 = __builtin_amdgcn_wmma_f32_16x16x32_bf16(false, a1, false, b1,
                                                           (short)0, acc1, false, false);
            acc2 = __builtin_amdgcn_wmma_f32_16x16x32_bf16(false, a2, false, b2,
                                                           (short)0, acc2, false, false);
        }
        #pragma unroll
        for (int r = 0; r < 8; ++r) {
            int m   = r + half * 8;
            int col = nt * 16 + ln;
            bool msk = mask[b * SEQ + col] != 0;
            float s1 = acc1[r] * 0.125f;   // 1/sqrt(64)
            float s2 = acc2[r] * 0.125f;
            if (msk) { s1 = -1e9f; s2 = -1e9f; }
            float sp = (s2 > 20.f) ? s2 : log1pf(expf(s2));     // softplus
            float kw = fminf(fmaxf(sp, 1e-2f), 1e3f);
            sc1[m][col] = s1;
            kwl[m][col] = (__bf16)kw;
            kout[(hbase + qbase + m) * SEQ + col] = kw;         // k_wei output (fp32)
        }
    }
    __syncthreads();

    // per-row softmax + Weibull reparameterized sample + normalize
    for (int rr = 0; rr < 2; ++rr) {
        int row = wv * 2 + rr;
        size_t rowg = (hbase + qbase + row) * SEQ;
        float sv[16];
        float mx = -3.0e38f;
        #pragma unroll
        for (int t = 0; t < 16; ++t) {
            sv[t] = sc1[row][lane + 32 * t];
            mx = fmaxf(mx, sv[t]);
        }
        #pragma unroll
        for (int o = 16; o > 0; o >>= 1) mx = fmaxf(mx, __shfl_xor(mx, o, 32));
        float sum = 0.f;
        #pragma unroll
        for (int t = 0; t < 16; ++t) { sv[t] = expf(sv[t] - mx); sum += sv[t]; }
        #pragma unroll
        for (int o = 16; o > 0; o >>= 1) sum += __shfl_xor(sum, o, 32);
        float inv = 1.f / sum;
        float rs = 0.f;
        float samp[16];
        #pragma unroll
        for (int t = 0; t < 16; ++t) {
            int col = lane + 32 * t;
            float ao   = sv[t] * inv;                       // att_ori
            float kw   = (float)kwl[row][col];
            float invk = 1.f / kw;
            float lam  = ao * expf(-lgammaf(1.f + invk));   // mean-matched scale
            float uu   = u[rowg + col];
            float x    = -logf(1.f - uu + EPSW) + EPSW;
            float s    = lam * expf(invk * logf(x));
            samp[t] = s;
            rs += s;
        }
        #pragma unroll
        for (int o = 16; o > 0; o >>= 1) rs += __shfl_xor(rs, o, 32);
        float rinv = 1.f / (rs + EPSW);
        #pragma unroll
        for (int t = 0; t < 16; ++t)
            att[rowg + lane + 32 * t] = (__bf16)(samp[t] * rinv);
    }
}

// ------------------------- att @ v WMMA GEMM -------------------------------
// per (b,h): [512x512]·[512x64]; double-buffered async V tile staging;
// output heads re-merged: atted[B,S,H,D]
__global__ void att_v_gemm(const __bf16* __restrict__ att, const __bf16* __restrict__ vh,
                           __bf16* __restrict__ atted) {
    __shared__ __bf16 Vt[2][32][HD];
    const int tid  = threadIdx.x, wv = tid >> 5, lane = tid & 31;
    const int half = lane >> 4, ln = lane & 15;
    const int bh = blockIdx.y, b = bh >> 4, h = bh & 15;
    const int qbase = blockIdx.x * 128 + wv * 16;
    const size_t hbase = (size_t)bh * SEQ;
    f32x8 acc[4];
    f32x8 z = {};
    #pragma unroll
    for (int t = 0; t < 4; ++t) acc[t] = z;

    // async staging: Vt 32x64 bf16 = 4KB -> 16B/thread (1x b128)
    const int vk = tid >> 3;            // 0..31
    const int vc = (tid & 7) * 8;       // 0..56 step 8
    const unsigned v_lds[2] = { lds_addr(&Vt[0][vk][vc]), lds_addr(&Vt[1][vk][vc]) };
    const __bf16* vgp = vh + (hbase + vk) * HD + vc;

    async_cp16(v_lds[0], gaddr(vgp));   // prologue: K-tile 0

    const __bf16* arow = att + (hbase + qbase + ln) * SEQ;
    const int NT = SEQ / 32;
    for (int t = 0; t < NT; ++t) {
        const int buf = t & 1;
        const int k0 = t * 32;
        if (t + 1 < NT) {
            async_cp16(v_lds[buf ^ 1], gaddr(vgp + (size_t)(k0 + 32) * HD));
            wait_async1();
        } else {
            wait_async0();
        }
        __syncthreads();

        bf16x16 a;
        #pragma unroll
        for (int i = 0; i < 16; ++i) a[i] = arow[k0 + kmap(i, half)];
        #pragma unroll
        for (int nt = 0; nt < 4; ++nt) {
            bf16x16 bb;
            #pragma unroll
            for (int i = 0; i < 16; ++i) bb[i] = Vt[buf][kmap(i, half)][nt * 16 + ln];
            acc[nt] = __builtin_amdgcn_wmma_f32_16x16x32_bf16(false, a, false, bb,
                                                              (short)0, acc[nt], false, false);
        }
        __syncthreads();
    }
    #pragma unroll
    for (int nt = 0; nt < 4; ++nt) {
        #pragma unroll
        for (int r = 0; r < 8; ++r) {
            int qrow = qbase + r + half * 8;
            int d    = nt * 16 + ln;
            atted[((size_t)(b * SEQ + qrow) * NH + h) * HD + d] = (__bf16)acc[nt][r];
        }
    }
}

// ---------------------------------------------------------------------------
extern "C" void kernel_launch(void* const* d_in, const int* in_sizes, int n_in,
                              void* d_out, int out_size, void* d_ws, size_t ws_size,
                              hipStream_t stream) {
    const float* v  = (const float*)d_in[0];
    const float* k  = (const float*)d_in[1];
    const float* q  = (const float*)d_in[2];
    const float* uw = (const float*)d_in[3];
    const unsigned char* mask = (const unsigned char*)d_in[4];
    const float* Wv  = (const float*)d_in[5];  const float* bv  = (const float*)d_in[6];
    const float* Wk  = (const float*)d_in[7];  const float* bk  = (const float*)d_in[8];
    const float* Wk2 = (const float*)d_in[9];  const float* bk2 = (const float*)d_in[10];
    const float* Wq  = (const float*)d_in[11]; const float* bq  = (const float*)d_in[12];
    const float* Wq2 = (const float*)d_in[13]; const float* bq2 = (const float*)d_in[14];
    const float* Wm  = (const float*)d_in[15]; const float* bm  = (const float*)d_in[16];

    float* out  = (float*)d_out;
    float* kout = out + (size_t)BATCH * SEQ * HID;   // k_wei, second tuple element

    char* ws = (char*)d_ws;
    size_t off = 0;
    auto alloc = [&](size_t bytes) -> void* {
        void* p = ws + off;
        off = (off + bytes + 255) & ~(size_t)255;
        return p;
    };
    const size_t NX = (size_t)BATCH * SEQ * HID;      // 2M
    const size_t NW = (size_t)HID * HID;              // 1M
    const size_t NA = (size_t)BATCH * NH * SEQ * SEQ; // 16.7M

    __bf16* qb   = (__bf16*)alloc(NX * 2);
    __bf16* kb   = (__bf16*)alloc(NX * 2);
    __bf16* vb   = (__bf16*)alloc(NX * 2);
    __bf16* Wvb  = (__bf16*)alloc(NW * 2);
    __bf16* Wkb  = (__bf16*)alloc(NW * 2);
    __bf16* Wk2b = (__bf16*)alloc(NW * 2);
    __bf16* Wqb  = (__bf16*)alloc(NW * 2);
    __bf16* Wq2b = (__bf16*)alloc(NW * 2);
    __bf16* Wmb  = (__bf16*)alloc(NW * 2);
    __bf16* vhb  = (__bf16*)alloc(NX * 2);
    __bf16* k1b  = (__bf16*)alloc(NX * 2);
    __bf16* q1b  = (__bf16*)alloc(NX * 2);
    __bf16* k2b  = (__bf16*)alloc(NX * 2);
    __bf16* q2b  = (__bf16*)alloc(NX * 2);
    __bf16* attb = (__bf16*)alloc(NA * 2);
    __bf16* attd = (__bf16*)alloc(NX * 2);

    // conversions
    cvt_f32_bf16<<<2048, 256, 0, stream>>>(q,  qb,  (int)NX);
    cvt_f32_bf16<<<2048, 256, 0, stream>>>(k,  kb,  (int)NX);
    cvt_f32_bf16<<<2048, 256, 0, stream>>>(v,  vb,  (int)NX);
    cvt_f32_bf16<<<1024, 256, 0, stream>>>(Wv,  Wvb,  (int)NW);
    cvt_f32_bf16<<<1024, 256, 0, stream>>>(Wk,  Wkb,  (int)NW);
    cvt_f32_bf16<<<1024, 256, 0, stream>>>(Wk2, Wk2b, (int)NW);
    cvt_f32_bf16<<<1024, 256, 0, stream>>>(Wq,  Wqb,  (int)NW);
    cvt_f32_bf16<<<1024, 256, 0, stream>>>(Wq2, Wq2b, (int)NW);
    cvt_f32_bf16<<<1024, 256, 0, stream>>>(Wm,  Wmb,  (int)NW);

    // projections -> head-split bf16 [B,H,S,D]
    dim3 ggrid(HID / 64, (BATCH * SEQ) / 128);
    gemm_wmma<0><<<ggrid, 256, 0, stream>>>(vb, Wvb,  bv,  vhb, nullptr);
    gemm_wmma<0><<<ggrid, 256, 0, stream>>>(kb, Wkb,  bk,  k1b, nullptr);
    gemm_wmma<0><<<ggrid, 256, 0, stream>>>(qb, Wqb,  bq,  q1b, nullptr);
    gemm_wmma<0><<<ggrid, 256, 0, stream>>>(kb, Wk2b, bk2, k2b, nullptr);
    gemm_wmma<0><<<ggrid, 256, 0, stream>>>(qb, Wq2b, bq2, q2b, nullptr);

    // fused attention scores + softmax + Weibull sampling
    dim3 agrid(SEQ / 16, BATCH * NH);
    attn_weibull<<<agrid, 256, 0, stream>>>(q1b, k1b, q2b, k2b, uw, mask, kout, attb);

    // att @ v -> [B,S,HID] bf16
    dim3 vgrid(SEQ / 128, BATCH * NH);
    att_v_gemm<<<vgrid, 256, 0, stream>>>(attb, vhb, attd);

    // final output projection -> fp32
    gemm_wmma<1><<<ggrid, 256, 0, stream>>>(attd, Wmb, bm, nullptr, out);
}
